// TransE_19756849561872
// MI455X (gfx1250) — matrix operand order repaired and checked by reading, not compiled, not run on hardware
//
#include <hip/hip_runtime.h>
#include <hip/hip_bf16.h>

typedef __attribute__((ext_vector_type(16))) _Float16 v16h;
typedef __attribute__((ext_vector_type(8)))  _Float16 v8h;
typedef __attribute__((ext_vector_type(4)))  _Float16 v4h;
typedef __attribute__((ext_vector_type(8)))  float    v8f;
typedef __attribute__((ext_vector_type(4)))  float    v4f;

#define HID 128
#define BT_STRIDE 136   // 128 + 8 halves pad: rows 16B-aligned, banks spread
#define A_STRIDE  132   // 128 + 4 halves pad: avoids 256B-stride bank conflicts
#define MARGIN 1.0f

// One block = one sample. 256 threads = 8 wave32.
__global__ void __launch_bounds__(256)
transr_score_kernel(const float* __restrict__ ent_emb,
                    const float* __restrict__ rel_emb,
                    const float* __restrict__ type_mat,
                    const int* __restrict__ pos_h,
                    const int* __restrict__ pos_t,
                    const int* __restrict__ pos_r,
                    const int* __restrict__ pos_type_r,
                    const int* __restrict__ neg_h,
                    const int* __restrict__ neg_t,
                    const int* __restrict__ neg_r,
                    float* __restrict__ ws)
{
    const int b   = blockIdx.x;
    const int tid = threadIdx.x;

    // LDS: transposed f16 T (sB[r][d]), f16 A (rows 0-3 = ph,pt,nh,nt), f32 results
    __shared__ __attribute__((aligned(16))) _Float16 sB[HID * BT_STRIDE]; // 34816 B
    __shared__ __attribute__((aligned(16))) _Float16 sA[16 * A_STRIDE];   //  4224 B
    __shared__ float sOut[4 * HID];                                       //  2048 B
    __shared__ float sPR[HID];
    __shared__ float sNR[HID];

    // ---------------- gather + stage vectors ----------------
    if (tid < HID) {
        const int r = tid;
        const float ph = ent_emb[(long)pos_h[b] * HID + r];
        const float pt = ent_emb[(long)pos_t[b] * HID + r];
        const float nh = ent_emb[(long)neg_h[b] * HID + r];
        const float nt = ent_emb[(long)neg_t[b] * HID + r];
        sA[0 * A_STRIDE + r] = (_Float16)ph;
        sA[1 * A_STRIDE + r] = (_Float16)pt;
        sA[2 * A_STRIDE + r] = (_Float16)nh;
        sA[3 * A_STRIDE + r] = (_Float16)nt;
        sPR[r] = rel_emb[(long)pos_r[b] * HID + r];
        sNR[r] = rel_emb[(long)neg_r[b] * HID + r];
    } else {
        // zero A rows 4..15 (12*128 halves / 128 threads = 12 each)
        const int t2 = tid - 128;
#pragma unroll
        for (int i = 0; i < 12; ++i) {
            const int idx = t2 * 12 + i;            // 0..1535
            const int row = 4 + (idx >> 7);
            const int col = idx & 127;
            sA[row * A_STRIDE + col] = (_Float16)0.0f;
        }
    }

    // ---------------- stage T: 4x4 register-tile transpose + f16 convert ----------
    // T row-major [d][r]; store sB[r][d]. 1024 4x4 tiles, 4 per thread.
    // One wave per iteration covers 4 full 512B rows -> coalesced b128 loads.
    const float* Tm = type_mat + (long)pos_type_r[b] * (HID * HID);
#pragma unroll
    for (int it = 0; it < 4; ++it) {
        const int u  = tid + 256 * it;              // 0..1023
        const int rb = (u & 31) * 4;                // column block (r)
        const int db = (u >> 5) * 4;                // row block (d)
        const float* src = Tm + db * HID + rb;
        const v4f row0 = *(const v4f*)(src);
        const v4f row1 = *(const v4f*)(src + HID);
        const v4f row2 = *(const v4f*)(src + 2 * HID);
        const v4f row3 = *(const v4f*)(src + 3 * HID);
#pragma unroll
        for (int i = 0; i < 4; ++i) {
            v4h q;                                   // packed cvt -> one b64 store
            q[0] = (_Float16)row0[i];
            q[1] = (_Float16)row1[i];
            q[2] = (_Float16)row2[i];
            q[3] = (_Float16)row3[i];
            *(v4h*)&sB[(rb + i) * BT_STRIDE + db] = q;
        }
    }
    __syncthreads();

    // ---------------- WMMA: [16x128] @ [128x128] -> rows 0..3 used ----------------
    const int wave = tid >> 5;        // 0..7 -> N-tile
    const int lane = tid & 31;
    const int grp  = lane >> 4;       // half-wave
    const int m    = lane & 15;
    const int n0   = wave * 16;

    v8f acc = {};
#pragma unroll
    for (int kk = 0; kk < HID; kk += 32) {
        // A fragment: 16-bit A 16x32 layout (lane m = row; K = 16*grp + element, contiguous)
        v16h a;
#pragma unroll
        for (int p = 0; p < 8; ++p) {
            const int kb = 2 * p + 16 * grp;
            a[2 * p]     = sA[m * A_STRIDE + kk + kb];
            a[2 * p + 1] = sA[m * A_STRIDE + kk + kb + 1];
        }
        // B fragment: lane's N = m, halves = 16 consecutive K (transposed LDS)
        const _Float16* bp = &sB[(n0 + m) * BT_STRIDE + kk + grp * 16];
        const v8h blo = *(const v8h*)(bp);          // ds_load_b128
        const v8h bhi = *(const v8h*)(bp + 8);      // ds_load_b128
        const v16h bf = __builtin_shufflevector(blo, bhi,
            0, 1, 2, 3, 4, 5, 6, 7, 8, 9, 10, 11, 12, 13, 14, 15);

        acc = __builtin_amdgcn_wmma_f32_16x16x32_f16(
            /*neg_a=*/false, a, /*neg_b=*/false, bf,
            /*c_mod=*/(short)0, acc, /*reuse_a=*/false, /*reuse_b=*/false);
    }

    // D rows 0..3 live in VGPRs 0..3 of lanes 0..15 (M = vgpr for grp 0)
    if (grp == 0) {
#pragma unroll
        for (int i = 0; i < 4; ++i)
            sOut[i * HID + n0 + m] = acc[i];
    }
    __syncthreads();

    // ---------------- normalize + margin score (wave 0) ----------------
    if (wave == 0) {
        float ssq[6] = {0.f, 0.f, 0.f, 0.f, 0.f, 0.f};
#pragma unroll
        for (int rr = 0; rr < 4; ++rr) {
            const int r = lane + 32 * rr;
            const float ph = sOut[0 * HID + r];
            const float pt = sOut[1 * HID + r];
            const float nh = sOut[2 * HID + r];
            const float nt = sOut[3 * HID + r];
            const float pr = sPR[r];
            const float nr = sNR[r];
            ssq[0] += ph * ph; ssq[1] += pt * pt; ssq[2] += nh * nh;
            ssq[3] += nt * nt; ssq[4] += pr * pr; ssq[5] += nr * nr;
        }
        float inv[6];
#pragma unroll
        for (int i = 0; i < 6; ++i) {
#pragma unroll
            for (int off = 16; off > 0; off >>= 1)
                ssq[i] += __shfl_down(ssq[i], off, 32);
            const float s = __shfl(ssq[i], 0, 32);
            inv[i] = rsqrtf(s + 1e-12f);
        }
        float sc = 0.f;
#pragma unroll
        for (int rr = 0; rr < 4; ++rr) {
            const int r = lane + 32 * rr;
            sc += fabsf(sOut[0 * HID + r] * inv[0] + sPR[r] * inv[4]
                        - sOut[1 * HID + r] * inv[1]);   // p contribution
            sc -= fabsf(sOut[2 * HID + r] * inv[2] + sNR[r] * inv[5]
                        - sOut[3 * HID + r] * inv[3]);   // n contribution
        }
#pragma unroll
        for (int off = 16; off > 0; off >>= 1)
            sc += __shfl_down(sc, off, 32);
        if (lane == 0)
            ws[b] = fmaxf(sc + MARGIN, 0.0f);
    }
}

__global__ void __launch_bounds__(256)
reduce_mean_kernel(const float* __restrict__ ws, float* __restrict__ out, int n)
{
    __shared__ float sm[256];
    float s = 0.f;
    for (int i = threadIdx.x; i < n; i += 256) s += ws[i];
    sm[threadIdx.x] = s;
    __syncthreads();
#pragma unroll
    for (int off = 128; off > 0; off >>= 1) {
        if ((int)threadIdx.x < off) sm[threadIdx.x] += sm[threadIdx.x + off];
        __syncthreads();
    }
    if (threadIdx.x == 0) out[0] = sm[0] / (float)n;
}

extern "C" void kernel_launch(void* const* d_in, const int* in_sizes, int n_in,
                              void* d_out, int out_size, void* d_ws, size_t ws_size,
                              hipStream_t stream) {
    const float* ent_emb    = (const float*)d_in[0];
    const float* rel_emb    = (const float*)d_in[1];
    const float* type_mat   = (const float*)d_in[2];
    const int*   pos_h      = (const int*)d_in[3];
    const int*   pos_t      = (const int*)d_in[4];
    const int*   pos_r      = (const int*)d_in[5];
    const int*   pos_type_r = (const int*)d_in[6];
    const int*   neg_h      = (const int*)d_in[7];
    const int*   neg_t      = (const int*)d_in[8];
    const int*   neg_r      = (const int*)d_in[9];

    const int B = in_sizes[3];                 // 8192
    float* ws = (float*)d_ws;                  // B floats of scratch (32 KB)

    transr_score_kernel<<<B, 256, 0, stream>>>(
        ent_emb, rel_emb, type_mat, pos_h, pos_t, pos_r, pos_type_r,
        neg_h, neg_t, neg_r, ws);

    reduce_mean_kernel<<<1, 256, 0, stream>>>(ws, (float*)d_out, B);
}
